// ConformerLayer_23132693856324
// MI455X (gfx1250) — matrix operand order repaired and checked by reading, not compiled, not run on hardware
//
#include <hip/hip_runtime.h>

// ---------------------------------------------------------------------------
// Conformer block on MI455X (gfx1250): all GEMMs via v_wmma_f32_16x16x32_bf16,
// GEMM operand tiles staged with global_load_async_to_lds_b128 (ASYNCcnt).
// ---------------------------------------------------------------------------

typedef __bf16 bf16;
typedef __attribute__((ext_vector_type(16))) __bf16 v16bf;
typedef __attribute__((ext_vector_type(8)))  float  v8f;

union Frag {              // one WMMA A/B operand: 16 bf16 per lane = 2 x 16B
    v16bf v;
    uint4 q[2];
};

static __device__ __forceinline__ v8f v8f_zero() {
    v8f z = {0.f,0.f,0.f,0.f,0.f,0.f,0.f,0.f};
    return z;
}
static __device__ __forceinline__ float sigmoidf_(float x) {
    return 1.0f / (1.0f + __expf(-x));
}
// async copy of 16B: LDS[lds+off] = MEM[gaddr+off]; per-lane, EXEC-masked
static __device__ __forceinline__ void async_cp16(unsigned lds, const void* g) {
    asm volatile("global_load_async_to_lds_b128 %0, %1, off"
                 :: "v"(lds), "v"((unsigned long long)(size_t)g) : "memory");
}
static __device__ __forceinline__ void async_cp16_off16(unsigned lds, const void* g) {
    asm volatile("global_load_async_to_lds_b128 %0, %1, off offset:16"
                 :: "v"(lds), "v"((unsigned long long)(size_t)g) : "memory");
}
static __device__ __forceinline__ void wait_async0() {
    asm volatile("s_wait_asynccnt 0x0" ::: "memory");
}

// ============================ generic WMMA GEMM ============================
// C(M,N) = A(M,K) * B(N,K)^T  (A,B bf16 row-major; weights are stored (out,in)
// exactly as the reference, so x @ W.T maps directly).
// Tile: 128x128x32, 256 threads = 8 waves in a 4x2 grid, wave tile 32x64.
enum { EPI_F32 = 0, EPI_SWISH_BF16 = 1, EPI_RES = 2, EPI_RES_HALF = 3 };

template <int EPI>
__global__ __launch_bounds__(256)
void gemm_wmma(const bf16* __restrict__ A, const bf16* __restrict__ Bw,
               const float* __restrict__ bias, const float* __restrict__ res,
               float* __restrict__ outF, bf16* __restrict__ outB,
               int Mdim, int Ndim, int Kdim)
{
    constexpr int BM = 128, BN = 128, BK = 32, PK = 40;   // PK keeps 16B align
    __shared__ __align__(16) bf16 As[2][BM][PK];
    __shared__ __align__(16) bf16 Bs[2][BN][PK];

    const int tid  = threadIdx.x;
    const int lane = tid & 31;
    const int w    = tid >> 5;
    const int wm   = w >> 1;          // 0..3  (32 rows each)
    const int wn   = w & 1;           // 0..1  (64 cols each)
    const int m0   = blockIdx.y * BM;
    const int n0   = blockIdx.x * BN;
    const int l15  = lane & 15;
    const int kh   = lane >> 4;

    v8f acc[2][4];
    for (int i = 0; i < 2; ++i)
        for (int j = 0; j < 4; ++j) acc[i][j] = v8f_zero();

    const int lr = tid >> 1;          // 0..127: row inside tile
    const int lc = (tid & 1) * 16;    // 0 or 16: 16 bf16 = 32B per thread

    const unsigned asBase = (unsigned)(size_t)(&As[0][0][0]);
    const unsigned bsBase = (unsigned)(size_t)(&Bs[0][0][0]);

    // async global -> LDS staging (32B per thread per matrix per k-tile)
    auto load_tile = [&](int bufi, int k0) {
        // A tile (guard M edge: pos-proj GEMM has M=1023; stale LDS rows only
        // feed C rows the epilogue never stores)
        int gr = m0 + lr;
        unsigned lA = asBase + (unsigned)((((bufi * BM) + lr) * PK + lc) * 2);
        if (gr < Mdim) {
            const bf16* g = A + (size_t)gr * Kdim + k0 + lc;
            async_cp16(lA, g);
            async_cp16_off16(lA, g);
        }
        // B tile (N always a multiple of 128 here)
        unsigned lB = bsBase + (unsigned)((((bufi * BN) + lr) * PK + lc) * 2);
        const bf16* gB = Bw + (size_t)(n0 + lr) * Kdim + k0 + lc;
        async_cp16(lB, gB);
        async_cp16_off16(lB, gB);
    };

    const int nk = Kdim / BK;
    int buf = 0;
    load_tile(0, 0);
    wait_async0();
    __syncthreads();

    for (int kt = 0; kt < nk; ++kt) {
        if (kt + 1 < nk) load_tile(buf ^ 1, (kt + 1) * BK);

        Frag a[2], b[4];
        // A frag: lanes 0-15 -> K{0..15}, lanes 16-31 -> K{16..31}
        for (int mi = 0; mi < 2; ++mi) {
            int row = wm * 32 + mi * 16 + l15;
            a[mi].q[0] = *reinterpret_cast<const uint4*>(&As[buf][row][16 * kh]);
            a[mi].q[1] = *reinterpret_cast<const uint4*>(&As[buf][row][16 * kh + 8]);
        }
        // B frag: lane holds 16 consecutive K of one N column
        for (int ni = 0; ni < 4; ++ni) {
            int row = wn * 64 + ni * 16 + l15;
            b[ni].q[0] = *reinterpret_cast<const uint4*>(&Bs[buf][row][16 * kh]);
            b[ni].q[1] = *reinterpret_cast<const uint4*>(&Bs[buf][row][16 * kh + 8]);
        }
        for (int mi = 0; mi < 2; ++mi)
            for (int ni = 0; ni < 4; ++ni)
                acc[mi][ni] = __builtin_amdgcn_wmma_f32_16x16x32_bf16(
                    false, a[mi].v, false, b[ni].v, (short)0, acc[mi][ni],
                    false, false);
        wait_async0();          // next tile landed (this wave's copies)
        __syncthreads();        // publish WGP-wide, and retire reads of buf
        buf ^= 1;
    }

    // epilogue:  C layout: lanes 0-15 -> M=r, lanes 16-31 -> M=r+8, N=lane&15
    for (int mi = 0; mi < 2; ++mi)
        for (int ni = 0; ni < 4; ++ni) {
            int gn = n0 + wn * 64 + ni * 16 + l15;
            float bv = bias ? bias[gn] : 0.0f;
            for (int r = 0; r < 8; ++r) {
                int gm = m0 + wm * 32 + mi * 16 + r + 8 * kh;
                if (gm >= Mdim) continue;
                float v = acc[mi][ni][r] + bv;
                size_t o = (size_t)gm * Ndim + gn;
                if (EPI == EPI_F32)            outF[o] = v;
                else if (EPI == EPI_SWISH_BF16) outB[o] = (bf16)(v * sigmoidf_(v));
                else if (EPI == EPI_RES)        outF[o] = res[o] + v;
                else /* EPI_RES_HALF */         outF[o] = res[o] + 0.5f * v;
            }
        }
}

// ====================== fused rel-pos flash attention ======================
// One wave per block, one (b, h, 16-query tile). Computes AC and shifted BD
// via WMMA, softmax in LDS, then attn @ V via WMMA. T=512, HD=64.
// K/V/P operands are L2-resident and reused across blocks -> read fragments
// straight from global (no LDS staging would reduce traffic).
__global__ __launch_bounds__(32)
void attn_k(const bf16* __restrict__ qu, const bf16* __restrict__ qv,
            const bf16* __restrict__ kk, const bf16* __restrict__ vT,
            const bf16* __restrict__ pb, bf16* __restrict__ outB)
{
    __shared__ __align__(16) float sS[16][520];   // scores / exp values
    __shared__ __align__(16) float sBD[16][36];   // BD gather staging
    __shared__ __align__(16) bf16  sP[16][512];   // probs (A operand)

    const int lane = threadIdx.x & 31;
    const int l15  = lane & 15;
    const int hi   = lane >> 4;
    const int i0   = blockIdx.x * 16;
    const int h    = blockIdx.y;
    const int b    = blockIdx.z;
    const int bh   = b * 8 + h;

    const bf16* quB = qu + ((size_t)bh * 512 + i0) * 64;
    const bf16* qvB = qv + ((size_t)bh * 512 + i0) * 64;
    const bf16* kB  = kk + (size_t)bh * 512 * 64;        // (t, hd)
    const bf16* vB  = vT + (size_t)bh * 64 * 512;        // (hd, t)
    const bf16* pB  = pb + (size_t)h * 1024 * 64;        // (2t-1 pad, hd)

    // Q fragments (A operand), 2 k-steps covering HD=64
    Frag au[2], av[2];
    for (int ks = 0; ks < 2; ++ks) {
        const bf16* a0 = quB + (size_t)l15 * 64 + ks * 32 + 16 * hi;
        au[ks].q[0] = *reinterpret_cast<const uint4*>(a0);
        au[ks].q[1] = *reinterpret_cast<const uint4*>(a0 + 8);
        const bf16* a1 = qvB + (size_t)l15 * 64 + ks * 32 + 16 * hi;
        av[ks].q[0] = *reinterpret_cast<const uint4*>(a1);
        av[ks].q[1] = *reinterpret_cast<const uint4*>(a1 + 8);
    }

    const float scale = 0.125f;   // HD^-0.5
    for (int j0 = 0; j0 < 512; j0 += 16) {
        // AC = (q+u) @ k^T for this 16x16 key tile
        v8f acc = v8f_zero();
        for (int ks = 0; ks < 2; ++ks) {
            Frag bk;
            const bf16* kp = kB + (size_t)(j0 + l15) * 64 + ks * 32 + 16 * hi;
            bk.q[0] = *reinterpret_cast<const uint4*>(kp);
            bk.q[1] = *reinterpret_cast<const uint4*>(kp + 8);
            acc = __builtin_amdgcn_wmma_f32_16x16x32_bf16(
                false, au[ks].v, false, bk.v, (short)0, acc, false, false);
        }
        // BD band: rows n0..n0+31 of P cover all n = 511 + j - i needed here
        int n0 = 511 + j0 - i0 - 15;                // always in [0, 992]
        v8f bd0 = v8f_zero(), bd1 = v8f_zero();
        for (int ks = 0; ks < 2; ++ks) {
            Frag bp0, bp1;
            const bf16* p0 = pB + (size_t)(n0 + l15) * 64 + ks * 32 + 16 * hi;
            const bf16* p1 = pB + (size_t)(n0 + 16 + l15) * 64 + ks * 32 + 16 * hi;
            bp0.q[0] = *reinterpret_cast<const uint4*>(p0);
            bp0.q[1] = *reinterpret_cast<const uint4*>(p0 + 8);
            bp1.q[0] = *reinterpret_cast<const uint4*>(p1);
            bp1.q[1] = *reinterpret_cast<const uint4*>(p1 + 8);
            bd0 = __builtin_amdgcn_wmma_f32_16x16x32_bf16(
                false, av[ks].v, false, bp0.v, (short)0, bd0, false, false);
            bd1 = __builtin_amdgcn_wmma_f32_16x16x32_bf16(
                false, av[ks].v, false, bp1.v, (short)0, bd1, false, false);
        }
        // stage BD tiles, then gather the rel-shift diagonal (d = jj-ii+15)
        for (int r = 0; r < 8; ++r) {
            sBD[r + 8 * hi][l15]      = bd0[r];
            sBD[r + 8 * hi][16 + l15] = bd1[r];
        }
        for (int r = 0; r < 8; ++r) {
            int ii = r + 8 * hi;
            int d  = l15 - ii + 15;                 // in [0, 30]
            sS[ii][j0 + l15] = (acc[r] + sBD[ii][d]) * scale;
        }
    }

    // softmax: 2 lanes per row (lane, lane+16), 256 cols each
    {
        int row = l15;
        int c0  = hi * 256;
        float mx = -1e30f;
        for (int j = 0; j < 256; ++j) mx = fmaxf(mx, sS[row][c0 + j]);
        mx = fmaxf(mx, __shfl_xor(mx, 16));
        float sum = 0.f;
        for (int j = 0; j < 256; ++j) {
            float e = __expf(sS[row][c0 + j] - mx);
            sS[row][c0 + j] = e;
            sum += e;
        }
        sum += __shfl_xor(sum, 16);
        float rs = 1.0f / sum;
        for (int j = 0; j < 256; ++j)
            sP[row][c0 + j] = (bf16)(sS[row][c0 + j] * rs);
    }

    // out = attn @ V   (V pre-transposed to (hd, t) so B frags are contiguous)
    v8f ao[4];
    for (int ni = 0; ni < 4; ++ni) ao[ni] = v8f_zero();
    for (int kt = 0; kt < 16; ++kt) {
        Frag ap;
        ap.q[0] = *reinterpret_cast<const uint4*>(&sP[l15][kt * 32 + 16 * hi]);
        ap.q[1] = *reinterpret_cast<const uint4*>(&sP[l15][kt * 32 + 16 * hi + 8]);
        for (int ni = 0; ni < 4; ++ni) {
            Frag bv_;
            const bf16* vp = vB + (size_t)(ni * 16 + l15) * 512 + kt * 32 + 16 * hi;
            bv_.q[0] = *reinterpret_cast<const uint4*>(vp);
            bv_.q[1] = *reinterpret_cast<const uint4*>(vp + 8);
            ao[ni] = __builtin_amdgcn_wmma_f32_16x16x32_bf16(
                false, ap.v, false, bv_.v, (short)0, ao[ni], false, false);
        }
    }
    // scatter to (t, b, d) bf16 for the out-proj GEMM
    for (int ni = 0; ni < 4; ++ni)
        for (int r = 0; r < 8; ++r) {
            int ti  = i0 + r + 8 * hi;
            int col = h * 64 + ni * 16 + l15;
            outB[((size_t)ti * 16 + b) * 512 + col] = (bf16)ao[ni][r];
        }
}

// ============================ small VALU kernels ===========================
__global__ __launch_bounds__(256)
void ln_k(const float* __restrict__ x, const float* __restrict__ g,
          const float* __restrict__ bb, bf16* __restrict__ outB,
          float* __restrict__ outF)
{
    const int lane = threadIdx.x & 31, wave = threadIdx.x >> 5;
    const size_t row = (size_t)blockIdx.x * 8 + wave;
    const float4* xr = reinterpret_cast<const float4*>(x + row * 512);
    float4 v[4];
    float s = 0.f;
    for (int i = 0; i < 4; ++i) {
        v[i] = xr[i * 32 + lane];
        s += v[i].x + v[i].y + v[i].z + v[i].w;
    }
    for (int o = 16; o; o >>= 1) s += __shfl_xor(s, o);
    float mean = s * (1.f / 512.f);
    float q = 0.f;
    for (int i = 0; i < 4; ++i) {
        float dx = v[i].x - mean, dy = v[i].y - mean;
        float dz = v[i].z - mean, dw = v[i].w - mean;
        q += dx * dx + dy * dy + dz * dz + dw * dw;
    }
    for (int o = 16; o; o >>= 1) q += __shfl_xor(q, o);
    float rstd = rsqrtf(q * (1.f / 512.f) + 1e-5f);
    const float4* g4 = reinterpret_cast<const float4*>(g);
    const float4* b4 = reinterpret_cast<const float4*>(bb);
    for (int i = 0; i < 4; ++i) {
        float4 gg = g4[i * 32 + lane], bv = b4[i * 32 + lane];
        float4 o;
        o.x = (v[i].x - mean) * rstd * gg.x + bv.x;
        o.y = (v[i].y - mean) * rstd * gg.y + bv.y;
        o.z = (v[i].z - mean) * rstd * gg.z + bv.z;
        o.w = (v[i].w - mean) * rstd * gg.w + bv.w;
        int col = (i * 32 + lane) * 4;
        if (outF)
            reinterpret_cast<float4*>(outF + row * 512)[i * 32 + lane] = o;
        if (outB) {
            bf16* ob = outB + row * 512 + col;
            ob[0] = (bf16)o.x; ob[1] = (bf16)o.y;
            ob[2] = (bf16)o.z; ob[3] = (bf16)o.w;
        }
    }
}

__global__ void cvt_k(const float* __restrict__ in, bf16* __restrict__ out, int n)
{
    int i = blockIdx.x * 256 + threadIdx.x;
    if (i < n) out[i] = (bf16)in[i];
}

__global__ void qkvsplit_k(const float* __restrict__ qkv,
                           const float* __restrict__ pu, const float* __restrict__ pv,
                           bf16* __restrict__ qu, bf16* __restrict__ qvb,
                           bf16* __restrict__ kk, bf16* __restrict__ vT)
{
    int idx = blockIdx.x * 256 + threadIdx.x;      // M*D
    int c = idx & 511, m = idx >> 9;
    int b = m & 15, t = m >> 4;
    int h = c >> 6, e = c & 63;
    size_t base = (size_t)m * 1536;
    float q = qkv[base + c], k = qkv[base + 512 + c], v = qkv[base + 1024 + c];
    size_t bh = (size_t)(b * 8 + h);
    size_t qi = (bh * 512 + t) * 64 + e;
    qu[qi]  = (bf16)(q + pu[c]);
    qvb[qi] = (bf16)(q + pv[c]);
    kk[qi]  = (bf16)k;
    vT[(bh * 64 + e) * 512 + t] = (bf16)v;
}

__global__ void posarr_k(const float* __restrict__ p, bf16* __restrict__ pb)
{
    int idx = blockIdx.x * 256 + threadIdx.x;      // 1024*512 (padded row 1023)
    int c = idx & 511, n = idx >> 9;
    int h = c >> 6, e = c & 63;
    float v = (n < 1023) ? p[(size_t)n * 512 + c] : 0.f;
    pb[((size_t)h * 1024 + n) * 64 + e] = (bf16)v;
}

__global__ void glu_k(const float* __restrict__ y, float* __restrict__ out)
{
    int idx = blockIdx.x * 256 + threadIdx.x;      // M*D
    int c = idx & 511, m = idx >> 9;
    float a = y[(size_t)m * 1024 + c];
    float g = y[(size_t)m * 1024 + 512 + c];
    out[idx] = a * sigmoidf_(g);
}

__global__ void dwconv_k(const float* __restrict__ x, const float* __restrict__ w,
                         const float* __restrict__ bias, float* __restrict__ out)
{
    int idx = blockIdx.x * 256 + threadIdx.x;      // M*D
    int c = idx & 511, m = idx >> 9;
    int b = m & 15, t = m >> 4;
    float acc = bias[c];
    for (int k = 0; k < 31; ++k) {
        int tt = t + k - 15;
        if (tt >= 0 && tt < 512)
            acc += x[(((size_t)tt * 16 + b) << 9) + c] * w[c * 31 + k];
    }
    out[idx] = acc;
}

__global__ __launch_bounds__(256)
void bnstats_k(const float* __restrict__ x, float* __restrict__ sum,
               float* __restrict__ sumsq)
{
    __shared__ float s1[256], s2[256];
    int c = blockIdx.x;
    float a = 0.f, q = 0.f;
    for (int m = threadIdx.x; m < 8192; m += 256) {
        float v = x[(size_t)m * 512 + c];
        a += v; q += v * v;
    }
    s1[threadIdx.x] = a; s2[threadIdx.x] = q;
    __syncthreads();
    for (int st = 128; st; st >>= 1) {
        if (threadIdx.x < (unsigned)st) {
            s1[threadIdx.x] += s1[threadIdx.x + st];
            s2[threadIdx.x] += s2[threadIdx.x + st];
        }
        __syncthreads();
    }
    if (threadIdx.x == 0) { sum[c] = s1[0]; sumsq[c] = s2[0]; }
}

__global__ void bnapply_k(const float* __restrict__ x, const float* __restrict__ sum,
                          const float* __restrict__ sumsq, const float* __restrict__ g,
                          const float* __restrict__ bb, bf16* __restrict__ out)
{
    int idx = blockIdx.x * 256 + threadIdx.x;      // M*D
    int c = idx & 511;
    float mean = sum[c] * (1.f / 8192.f);
    float var  = sumsq[c] * (1.f / 8192.f) - mean * mean;
    float v = (x[idx] - mean) * rsqrtf(var + 1e-5f) * g[c] + bb[c];
    out[idx] = (bf16)(v * sigmoidf_(v));
}

// ================================ launcher =================================
extern "C" void kernel_launch(void* const* d_in, const int* in_sizes, int n_in,
                              void* d_out, int out_size, void* d_ws, size_t ws_size,
                              hipStream_t stream)
{
    (void)in_sizes; (void)n_in; (void)out_size; (void)ws_size;
    const float* x        = (const float*)d_in[0];
    const float* pos_emb  = (const float*)d_in[1];
    const float* in_w     = (const float*)d_in[2];
    const float* in_b     = (const float*)d_in[3];
    const float* out_w    = (const float*)d_in[4];
    const float* out_b    = (const float*)d_in[5];
    const float* pos_w    = (const float*)d_in[6];
    const float* pbu      = (const float*)d_in[7];
    const float* pbv      = (const float*)d_in[8];
    const float* ffm_w1   = (const float*)d_in[9];
    const float* ffm_b1   = (const float*)d_in[10];
    const float* ffm_w2   = (const float*)d_in[11];
    const float* ffm_b2   = (const float*)d_in[12];
    const float* ff_w1    = (const float*)d_in[13];
    const float* ff_b1    = (const float*)d_in[14];
    const float* ff_w2    = (const float*)d_in[15];
    const float* ff_b2    = (const float*)d_in[16];
    const float* pw1_w    = (const float*)d_in[17];
    const float* pw1_b    = (const float*)d_in[18];
    const float* dw_w     = (const float*)d_in[19];
    const float* dw_b     = (const float*)d_in[20];
    const float* bn_g     = (const float*)d_in[21];
    const float* bn_b     = (const float*)d_in[22];
    const float* pw2_w    = (const float*)d_in[23];
    const float* pw2_b    = (const float*)d_in[24];
    const float* ln_mac_g = (const float*)d_in[25];
    const float* ln_mac_b = (const float*)d_in[26];
    const float* ln_mha_g = (const float*)d_in[27];
    const float* ln_mha_b = (const float*)d_in[28];
    const float* ln_conv_g= (const float*)d_in[29];
    const float* ln_conv_b= (const float*)d_in[30];
    const float* ln_ff_g  = (const float*)d_in[31];
    const float* ln_ff_b  = (const float*)d_in[32];
    const float* ln_fin_g = (const float*)d_in[33];
    const float* ln_fin_b = (const float*)d_in[34];
    float* out = (float*)d_out;

    const int M = 8192, D = 512, FF = 2048, TP = 1023;

    char* wsb = (char*)d_ws;
    size_t off = 0;
    auto alloc = [&](size_t bytes) -> char* {
        char* p = wsb + off;
        off += (bytes + 255) & ~(size_t)255;
        return p;
    };

    // bf16 weights
    bf16* ffm_w1_bf = (bf16*)alloc((size_t)FF * D * 2);
    bf16* ffm_w2_bf = (bf16*)alloc((size_t)D * FF * 2);
    bf16* ff_w1_bf  = (bf16*)alloc((size_t)FF * D * 2);
    bf16* ff_w2_bf  = (bf16*)alloc((size_t)D * FF * 2);
    bf16* in_w_bf   = (bf16*)alloc((size_t)3 * D * D * 2);
    bf16* out_w_bf  = (bf16*)alloc((size_t)D * D * 2);
    bf16* pos_w_bf  = (bf16*)alloc((size_t)D * D * 2);
    bf16* pw1_w_bf  = (bf16*)alloc((size_t)2 * D * D * 2);
    bf16* pw2_w_bf  = (bf16*)alloc((size_t)D * D * 2);
    bf16* posemb_bf = (bf16*)alloc((size_t)TP * D * 2);
    // activations
    bf16*  xn_bf  = (bf16*)alloc((size_t)M * D * 2);
    float* xn_f   = (float*)alloc((size_t)M * D * 4);
    float* x1     = (float*)alloc((size_t)M * D * 4);
    float* x2     = (float*)alloc((size_t)M * D * 4);
    float* x3     = (float*)alloc((size_t)M * D * 4);
    float* x4     = (float*)alloc((size_t)M * D * 4);
    char*  U1     = alloc((size_t)M * 3 * D * 4);     // qkv f32 / FF hidden / pw1 out
    bf16*  qu_bf  = (bf16*)alloc((size_t)M * D * 2);
    bf16*  qv_bf  = (bf16*)alloc((size_t)M * D * 2);
    bf16*  k_bf   = (bf16*)alloc((size_t)M * D * 2);
    bf16*  vT_bf  = (bf16*)alloc((size_t)M * D * 2);
    float* p_f    = (float*)alloc((size_t)TP * D * 4);
    bf16*  p_bf   = (bf16*)alloc((size_t)8 * 1024 * 64 * 2);
    bf16*  att_bf = (bf16*)alloc((size_t)M * D * 2);
    float* glu_f  = (float*)alloc((size_t)M * D * 4);
    float* dw_f   = (float*)alloc((size_t)M * D * 4);
    bf16*  cv_bf  = (bf16*)alloc((size_t)M * D * 2);
    float* bsum   = (float*)alloc(512 * 4);
    float* bsq    = (float*)alloc(512 * 4);

    float* qkv_f  = (float*)U1;
    bf16*  hid_bf = (bf16*)U1;
    float* y_f    = (float*)U1;

    auto cvt = [&](const float* s, bf16* d, int n) {
        cvt_k<<<dim3((n + 255) / 256), 256, 0, stream>>>(s, d, n);
    };
    cvt(ffm_w1, ffm_w1_bf, FF * D);
    cvt(ffm_w2, ffm_w2_bf, D * FF);
    cvt(ff_w1,  ff_w1_bf,  FF * D);
    cvt(ff_w2,  ff_w2_bf,  D * FF);
    cvt(in_w,   in_w_bf,   3 * D * D);
    cvt(out_w,  out_w_bf,  D * D);
    cvt(pos_w,  pos_w_bf,  D * D);
    cvt(pw1_w,  pw1_w_bf,  2 * D * D);
    cvt(pw2_w,  pw2_w_bf,  D * D);
    cvt(pos_emb, posemb_bf, TP * D);

    const dim3 blk(256);
    const dim3 lgrid(M / 8);
    const int  EW = M * D / 256;      // elementwise grid

    // ---- macaron FF: x1 = xn + 0.5*(swish(xn W1^T + b1) W2^T + b2) ----
    ln_k<<<lgrid, blk, 0, stream>>>(x, ln_mac_g, ln_mac_b, xn_bf, xn_f);
    gemm_wmma<EPI_SWISH_BF16><<<dim3(FF / 128, M / 128), blk, 0, stream>>>(
        xn_bf, ffm_w1_bf, ffm_b1, nullptr, nullptr, hid_bf, M, FF, D);
    gemm_wmma<EPI_RES_HALF><<<dim3(D / 128, M / 128), blk, 0, stream>>>(
        hid_bf, ffm_w2_bf, ffm_b2, xn_f, x1, nullptr, M, D, FF);

    // ---- rel-pos MHA ----
    ln_k<<<lgrid, blk, 0, stream>>>(x1, ln_mha_g, ln_mha_b, xn_bf, nullptr);
    gemm_wmma<EPI_F32><<<dim3(3 * D / 128, M / 128), blk, 0, stream>>>(
        xn_bf, in_w_bf, in_b, nullptr, qkv_f, nullptr, M, 3 * D, D);
    gemm_wmma<EPI_F32><<<dim3(D / 128, (TP + 127) / 128), blk, 0, stream>>>(
        posemb_bf, pos_w_bf, nullptr, nullptr, p_f, nullptr, TP, D, D);
    qkvsplit_k<<<dim3(EW), blk, 0, stream>>>(qkv_f, pbu, pbv,
                                             qu_bf, qv_bf, k_bf, vT_bf);
    posarr_k<<<dim3(1024 * 512 / 256), blk, 0, stream>>>(p_f, p_bf);
    attn_k<<<dim3(32, 8, 16), dim3(32), 0, stream>>>(
        qu_bf, qv_bf, k_bf, vT_bf, p_bf, att_bf);
    gemm_wmma<EPI_RES><<<dim3(D / 128, M / 128), blk, 0, stream>>>(
        att_bf, out_w_bf, out_b, x1, x2, nullptr, M, D, D);

    // ---- conv module ----
    ln_k<<<lgrid, blk, 0, stream>>>(x2, ln_conv_g, ln_conv_b, xn_bf, nullptr);
    gemm_wmma<EPI_F32><<<dim3(2 * D / 128, M / 128), blk, 0, stream>>>(
        xn_bf, pw1_w_bf, pw1_b, nullptr, y_f, nullptr, M, 2 * D, D);
    glu_k<<<dim3(EW), blk, 0, stream>>>(y_f, glu_f);
    dwconv_k<<<dim3(EW), blk, 0, stream>>>(glu_f, dw_w, dw_b, dw_f);
    bnstats_k<<<dim3(512), blk, 0, stream>>>(dw_f, bsum, bsq);
    bnapply_k<<<dim3(EW), blk, 0, stream>>>(dw_f, bsum, bsq, bn_g, bn_b, cv_bf);
    gemm_wmma<EPI_RES><<<dim3(D / 128, M / 128), blk, 0, stream>>>(
        cv_bf, pw2_w_bf, pw2_b, x2, x3, nullptr, M, D, D);

    // ---- final FF + final LN ----
    ln_k<<<lgrid, blk, 0, stream>>>(x3, ln_ff_g, ln_ff_b, xn_bf, nullptr);
    gemm_wmma<EPI_SWISH_BF16><<<dim3(FF / 128, M / 128), blk, 0, stream>>>(
        xn_bf, ff_w1_bf, ff_b1, nullptr, nullptr, hid_bf, M, FF, D);
    gemm_wmma<EPI_RES_HALF><<<dim3(D / 128, M / 128), blk, 0, stream>>>(
        hid_bf, ff_w2_bf, ff_b2, x3, x4, nullptr, M, D, FF);
    ln_k<<<lgrid, blk, 0, stream>>>(x4, ln_fin_g, ln_fin_b, nullptr, out);
}
